// MultiTaskMLP_70179765617302
// MI455X (gfx1250) — matrix operand order repaired and checked by reading, not compiled, not run on hardware
//
#include <hip/hip_runtime.h>
#include <hip/hip_bf16.h>
#include <stdint.h>

#define B_   4096
#define T_   128
#define IN_  768
#define H_   256
#define BM   32          // batch rows per block
#define H1P  (H_ + 8)    // LDS pitch (264 elems = 528B, 4-bank skew per row)

typedef __attribute__((ext_vector_type(16))) __bf16 v16bf;
typedef __attribute__((ext_vector_type(8)))  float  v8f;

union FragU { v16bf v; uint4 q[2]; };

// ---- WMMA A fragment from row-major bf16 [rows][pitch], tile at (row0, k0) ----
// ISA 16-bit A 16x32 layout: lanes 0-15 row M=lane, VGPR0-3 K=0..7, VGPR4-7 K=16..23;
// lanes 16-31 same rows, K=8..15 and 24..31.
__device__ __forceinline__ v16bf load_a_frag(const __bf16* base, int pitch,
                                             int row0, int k0, int lane) {
  const int r  = row0 + (lane & 15);
  const int kh = (lane >> 4) << 3;          // 0 or 8
  const __bf16* p = base + (size_t)r * pitch + (k0 + kh);
  FragU f;
  f.q[0] = *(const uint4*)(p);              // K = k0+kh   .. +7
  f.q[1] = *(const uint4*)(p + 16);         // K = k0+kh+16.. +23
  return f.v;
}

// ---- WMMA B fragment from N-major bf16 weights [N][ldk], tile at (ncol0, k0) ----
// Layout: lanes 0-15 col N=lane hold K=k0..k0+7, k0+16..k0+23 (packed pairs per VGPR),
// lanes 16-31 hold K=k0+8..k0+15, k0+24..k0+31.
__device__ __forceinline__ v16bf load_b_frag(const __bf16* baseT, int ldk,
                                             int ncol0, int k0, int lane) {
  const int n  = ncol0 + (lane & 15);
  const int ks = k0 + ((lane >> 4) << 3);   // +0 or +8
  const __bf16* p = baseT + (size_t)n * ldk + ks;
  FragU f;
  f.q[0] = *(const uint4*)(p);              // K = ks   .. ks+7
  f.q[1] = *(const uint4*)(p + 16);         // K = ks+16.. ks+23
  return f.v;
}

__device__ __forceinline__ v8f wmma_bf16(v16bf a, v16bf b, v8f c) {
  return __builtin_amdgcn_wmma_f32_16x16x32_bf16(false, a, false, b,
                                                 (short)0, c, false, false);
}

// ------------------- pre-pass: fp32 -> bf16 elementwise -------------------
__global__ void k_cvt_bf16(const float* __restrict__ src, __bf16* __restrict__ dst, int n) {
  int i = blockIdx.x * blockDim.x + threadIdx.x;
  if (i < n) dst[i] = (__bf16)src[i];
}

// ---- pre-pass: per-task transpose fp32 [T][K][N] -> bf16 [T][N][K] (LDS tiled) ----
__global__ void k_transpose_bf16(const float* __restrict__ src, __bf16* __restrict__ dst,
                                 int K, int N) {
  __shared__ float tile[32][33];
  const int t  = blockIdx.z;
  const int k0 = blockIdx.x * 32;
  const int n0 = blockIdx.y * 32;
  const float* s = src + (size_t)t * K * N;
  __bf16*      d = dst + (size_t)t * K * N;
  const int tx = threadIdx.x, ty = threadIdx.y;
#pragma unroll
  for (int j = 0; j < 32; j += 8)
    tile[ty + j][tx] = s[(size_t)(k0 + ty + j) * N + (n0 + tx)];
  __syncthreads();
#pragma unroll
  for (int j = 0; j < 32; j += 8)
    d[(size_t)(n0 + ty + j) * K + (k0 + tx)] = (__bf16)tile[tx][ty + j];
}

// ------------------- fused 3-layer per-task MLP -------------------
__global__ __launch_bounds__(256)
void k_mlp(const __bf16* __restrict__ xbf, const __bf16* __restrict__ W1T,
           const __bf16* __restrict__ W2T, const float* __restrict__ b1,
           const float* __restrict__ b2,  const float* __restrict__ W3,
           const float* __restrict__ b3,  float* __restrict__ out) {
  const int t    = blockIdx.y;
  const int bm   = blockIdx.x;
  const int lane = threadIdx.x & 31;
  const int w    = threadIdx.x >> 5;      // wave 0..7
  const int n0   = w * 32;                // this wave's H columns [n0, n0+32)
  const int ln   = lane & 15;
  const int lh   = lane >> 4;             // 0/1

  __shared__ __align__(16) __bf16 h1s[BM][H1P];
  __shared__ float red[BM];
  if (threadIdx.x < BM) red[threadIdx.x] = 0.f;

  const __bf16* xrow = xbf + (size_t)(bm * BM) * IN_;
  const __bf16* w1   = W1T + (size_t)t * H_ * IN_;
  const __bf16* w2   = W2T + (size_t)t * H_ * H_;

  // ---------------- layer 1: h1 = relu(x @ W1 + b1) ----------------
  // Software-pipelined: fragments for step k+1 are loaded before step-k WMMAs,
  // so the 4 WMMAs per step overlap a full iteration of load latency.
  v8f acc[2][2] = {};                     // [m-tile][n-tile]
  v16bf a0 = load_a_frag(xrow, IN_,  0, 0, lane);
  v16bf a1 = load_a_frag(xrow, IN_, 16, 0, lane);
  v16bf c0 = load_b_frag(w1, IN_, n0,      0, lane);
  v16bf c1 = load_b_frag(w1, IN_, n0 + 16, 0, lane);
#pragma unroll
  for (int kb = 0; kb < IN_ - 32; kb += 32) {
    const int kn = kb + 32;
    v16bf a0n = load_a_frag(xrow, IN_,  0, kn, lane);
    v16bf a1n = load_a_frag(xrow, IN_, 16, kn, lane);
    v16bf c0n = load_b_frag(w1, IN_, n0,      kn, lane);
    v16bf c1n = load_b_frag(w1, IN_, n0 + 16, kn, lane);
    __builtin_prefetch((const void*)(w1 + (size_t)(n0 + ln) * IN_ + kn + 64), 0, 3);
    acc[0][0] = wmma_bf16(a0, c0, acc[0][0]);
    acc[1][0] = wmma_bf16(a1, c0, acc[1][0]);
    acc[0][1] = wmma_bf16(a0, c1, acc[0][1]);
    acc[1][1] = wmma_bf16(a1, c1, acc[1][1]);
    a0 = a0n; a1 = a1n; c0 = c0n; c1 = c1n;
  }
  acc[0][0] = wmma_bf16(a0, c0, acc[0][0]);
  acc[1][0] = wmma_bf16(a1, c0, acc[1][0]);
  acc[0][1] = wmma_bf16(a0, c1, acc[0][1]);
  acc[1][1] = wmma_bf16(a1, c1, acc[1][1]);

  // bias + relu -> bf16 -> LDS (C layout: VGPR r = row r / r+8 by lane half, col = ln)
#pragma unroll
  for (int nt = 0; nt < 2; ++nt) {
    const int nc = n0 + nt * 16 + ln;
    const float bias = b1[(size_t)t * H_ + nc];
#pragma unroll
    for (int mt = 0; mt < 2; ++mt) {
#pragma unroll
      for (int r = 0; r < 8; ++r) {
        const int m = mt * 16 + r + (lh << 3);
        h1s[m][nc] = (__bf16)fmaxf(acc[mt][nt][r] + bias, 0.f);
      }
    }
  }
  __syncthreads();

  // ---------------- layer 2: h2 = relu(h1 @ W2 + b2) ----------------
  v8f acc2[2][2] = {};
  const __bf16* h1p = &h1s[0][0];
  a0 = load_a_frag(h1p, H1P,  0, 0, lane);            // ds_load_b128 path
  a1 = load_a_frag(h1p, H1P, 16, 0, lane);
  c0 = load_b_frag(w2, H_, n0,      0, lane);
  c1 = load_b_frag(w2, H_, n0 + 16, 0, lane);
#pragma unroll
  for (int kb = 0; kb < H_ - 32; kb += 32) {
    const int kn = kb + 32;
    v16bf a0n = load_a_frag(h1p, H1P,  0, kn, lane);
    v16bf a1n = load_a_frag(h1p, H1P, 16, kn, lane);
    v16bf c0n = load_b_frag(w2, H_, n0,      kn, lane);
    v16bf c1n = load_b_frag(w2, H_, n0 + 16, kn, lane);
    __builtin_prefetch((const void*)(w2 + (size_t)(n0 + ln) * H_ + kn + 64), 0, 3);
    acc2[0][0] = wmma_bf16(a0, c0, acc2[0][0]);
    acc2[1][0] = wmma_bf16(a1, c0, acc2[1][0]);
    acc2[0][1] = wmma_bf16(a0, c1, acc2[0][1]);
    acc2[1][1] = wmma_bf16(a1, c1, acc2[1][1]);
    a0 = a0n; a1 = a1n; c0 = c0n; c1 = c1n;
  }
  acc2[0][0] = wmma_bf16(a0, c0, acc2[0][0]);
  acc2[1][0] = wmma_bf16(a1, c0, acc2[1][0]);
  acc2[0][1] = wmma_bf16(a0, c1, acc2[0][1]);
  acc2[1][1] = wmma_bf16(a1, c1, acc2[1][1]);

  // ---------------- layer 3: out = relu(h2) . W3 + b3 ----------------
  const float bias20 = b2[(size_t)t * H_ + n0 + ln];
  const float bias21 = b2[(size_t)t * H_ + n0 + 16 + ln];
  const float w30    = W3[(size_t)t * H_ + n0 + ln];
  const float w31    = W3[(size_t)t * H_ + n0 + 16 + ln];
#pragma unroll
  for (int mt = 0; mt < 2; ++mt) {
#pragma unroll
    for (int r = 0; r < 8; ++r) {
      const int m = mt * 16 + r + (lh << 3);
      const float p = fmaxf(acc2[mt][0][r] + bias20, 0.f) * w30 +
                      fmaxf(acc2[mt][1][r] + bias21, 0.f) * w31;
      atomicAdd(&red[m], p);                       // ds_add_f32
    }
  }
  __syncthreads();
  if (threadIdx.x < BM) {
    out[(size_t)(bm * BM + threadIdx.x) * T_ + t] = red[threadIdx.x] + b3[t];
  }
}

// ------------------------------------------------------------------
extern "C" void kernel_launch(void* const* d_in, const int* in_sizes, int n_in,
                              void* d_out, int out_size, void* d_ws, size_t ws_size,
                              hipStream_t stream) {
  const float* x  = (const float*)d_in[0];
  const float* W1 = (const float*)d_in[1];
  const float* b1 = (const float*)d_in[2];
  const float* W2 = (const float*)d_in[3];
  const float* b2 = (const float*)d_in[4];
  const float* W3 = (const float*)d_in[5];
  const float* b3 = (const float*)d_in[6];
  float* out = (float*)d_out;

  char* ws = (char*)d_ws;
  size_t off = 0;
  __bf16* xbf = (__bf16*)(ws + off);
  off += (size_t)B_ * IN_ * sizeof(__bf16);        off = (off + 255) & ~(size_t)255;
  __bf16* W1T = (__bf16*)(ws + off);
  off += (size_t)T_ * H_ * IN_ * sizeof(__bf16);   off = (off + 255) & ~(size_t)255;
  __bf16* W2T = (__bf16*)(ws + off);

  // pre-pass: bf16 conversion + per-task weight transposes (memory-bound, ~10 us)
  k_cvt_bf16<<<(B_ * IN_ + 255) / 256, 256, 0, stream>>>(x, xbf, B_ * IN_);
  k_transpose_bf16<<<dim3(IN_ / 32, H_ / 32, T_), dim3(32, 8), 0, stream>>>(W1, W1T, IN_, H_);
  k_transpose_bf16<<<dim3(H_  / 32, H_ / 32, T_), dim3(32, 8), 0, stream>>>(W2, W2T, H_, H_);

  // fused MLP: grid (B/32 batch tiles, T tasks), 8 waves per block
  k_mlp<<<dim3(B_ / BM, T_), 256, 0, stream>>>(xbf, W1T, W2T, b1, b2, W3, b3, out);
}
